// BatchdenseGAT_5171140625171
// MI455X (gfx1250) — matrix operand order repaired and checked by reading, not compiled, not run on hardware
//
#include <hip/hip_runtime.h>
#include <cstdint>
#include <cstddef>

// ---- types for WMMA fragments (wave32, v_wmma_f32_16x16x32_f16) ----
typedef _Float16 h2   __attribute__((ext_vector_type(2)));
typedef _Float16 v16h __attribute__((ext_vector_type(16)));
typedef float    v8f  __attribute__((ext_vector_type(8)));

union AF { v16h v; h2 p[8]; };
union CF { v8f  v; float f[8]; };

#define B_    16
#define N_    512
#define F0_   64
#define DE_   64
#define UE_   3
#define K0V_  131      // valid K of layer 0 (64+64+3)
#define K0P_  160      // padded to multiple of 32
#define H_    8
#define O_    128
#define NOUT_ 1024     // H*O
#define K1_   1024

// A-fragment 16x32 f16: lane lm = M row; k = (v>>2)*16 + hi*8 + (v&3)*2
__device__ __forceinline__ void load_afrag(const _Float16* __restrict__ Arow,
                                           int kb, int hi, AF& a)
{
#pragma unroll
    for (int v = 0; v < 8; ++v) {
        int k = kb + ((v >> 2) << 4) + (hi << 3) + ((v & 3) << 1);
        a.p[v] = *(const h2*)(Arow + k);
    }
}
// B-fragment 32x16 f16: lane lm = N col; k = (v>>2)*16 + hi*8 + (v&3)*2
__device__ __forceinline__ void load_bfrag(const _Float16* __restrict__ Bcol,
                                           int kb, int hi, AF& bf)
{
#pragma unroll
    for (int v = 0; v < 8; ++v) {
        int k = kb + ((v >> 2) << 4) + (hi << 3) + ((v & 3) << 1);
        bf.p[v] = *(const h2*)(Bcol + k);
    }
}

// ---------------------------------------------------------------------------
// 1a. copy raw node features into packed f16 X, zero the K padding
// ---------------------------------------------------------------------------
__global__ __launch_bounds__(256) void pack_xh_kernel(
    const float* __restrict__ hfeat, _Float16* __restrict__ XA)
{
    size_t idx = (size_t)blockIdx.x * 256 + threadIdx.x;
    if (idx >= (size_t)B_ * N_ * K0P_) return;
    int    k  = (int)(idx % K0P_);
    size_t bn = idx / K0P_;
    if (k < F0_)                   XA[idx] = (_Float16)hfeat[bn * F0_ + k];
    else if (k >= F0_ + DE_ + UE_) XA[idx] = (_Float16)0.f;
    // cols [64,131) written by norm_pack_kernel
}

// ---------------------------------------------------------------------------
// 1b. gather + InstanceNorm over node dim per (batch, channel), write f16
// ---------------------------------------------------------------------------
__global__ __launch_bounds__(256) void norm_pack_kernel(
    const int* __restrict__ vertices, const float* __restrict__ user_emb,
    const float* __restrict__ emb_table,
    const float* __restrict__ n1w, const float* __restrict__ n1b,
    const float* __restrict__ n2w, const float* __restrict__ n2b,
    _Float16* __restrict__ XA)
{
    int b   = blockIdx.x / (DE_ + UE_);
    int c   = blockIdx.x % (DE_ + UE_);
    int tid = threadIdx.x;
    int n0 = tid, n1 = tid + 256;
    float v0, v1, gamma, beta; int col;
    if (c < DE_) {
        int t0 = vertices[b * N_ + n0], t1 = vertices[b * N_ + n1];
        v0 = emb_table[(size_t)t0 * DE_ + c];
        v1 = emb_table[(size_t)t1 * DE_ + c];
        gamma = n1w[c]; beta = n1b[c]; col = F0_ + c;
    } else {
        int cu = c - DE_;
        v0 = user_emb[((size_t)b * N_ + n0) * UE_ + cu];
        v1 = user_emb[((size_t)b * N_ + n1) * UE_ + cu];
        gamma = n2w[cu]; beta = n2b[cu]; col = F0_ + DE_ + cu;
    }
    __shared__ float rs[256], rq[256];
    rs[tid] = v0 + v1;
    rq[tid] = v0 * v0 + v1 * v1;
    __syncthreads();
    for (int s = 128; s > 0; s >>= 1) {
        if (tid < s) { rs[tid] += rs[tid + s]; rq[tid] += rq[tid + s]; }
        __syncthreads();
    }
    float mean = rs[0] * (1.0f / N_);
    float var  = rq[0] * (1.0f / N_) - mean * mean;   // biased, as jnp.var
    float rstd = rsqrtf(var + 1e-5f);
    XA[((size_t)b * N_ + n0) * K0P_ + col] = (_Float16)((v0 - mean) * rstd * gamma + beta);
    XA[((size_t)b * N_ + n1) * K0P_ + col] = (_Float16)((v1 - mean) * rstd * gamma + beta);
}

// ---------------------------------------------------------------------------
// 1c. weights [H,Kv,O] fp32 -> WT [NOUT][Kpad] f16 (n-major, contiguous k)
// ---------------------------------------------------------------------------
__global__ __launch_bounds__(256) void pack_w_kernel(
    const float* __restrict__ w, _Float16* __restrict__ WT, int K, int Kvalid)
{
    size_t idx = (size_t)blockIdx.x * 256 + threadIdx.x;
    if (idx >= (size_t)NOUT_ * K) return;
    int k  = (int)(idx % K);
    int nn = (int)(idx / K);
    int hh = nn >> 7, o = nn & 127;
    WT[idx] = (k < Kvalid) ? (_Float16)w[((size_t)hh * Kvalid + k) * O_ + o]
                           : (_Float16)0.f;
}

// ---------------------------------------------------------------------------
// 2. projection GEMM, software-pipelined (ping-pong fragment buffers):
//    stage s+1 fragment loads are issued BEFORE stage-s WMMAs so the wave
//    never has to drain loadcnt to 0 before the matrix op.
// ---------------------------------------------------------------------------
template<int K>
__global__ __launch_bounds__(256) void gemm_kernel(
    const _Float16* __restrict__ X, const _Float16* __restrict__ WT,
    float* __restrict__ HP, _Float16* __restrict__ HPT)
{
    constexpr int S = K / 32;
    int b    = blockIdx.z;
    int tid  = threadIdx.x;
    int wave = tid >> 5, lane = tid & 31;
    int lm = lane & 15, hi = lane >> 4;
    int m_base = blockIdx.x * 64  + (wave & 3) * 16;
    int n_wave = blockIdx.y * 128 + (wave >> 2) * 64;
    const _Float16* Arow = X + ((size_t)b * N_ + m_base + lm) * K;
    const _Float16* Bcol[4];
#pragma unroll
    for (int j = 0; j < 4; ++j)
        Bcol[j] = WT + (size_t)(n_wave + j * 16 + lm) * K;

    CF acc[4];
#pragma unroll
    for (int j = 0; j < 4; ++j)
#pragma unroll
        for (int q = 0; q < 8; ++q) acc[j].f[q] = 0.f;

    AF aP, bP[4], aQ, bQ[4];
    load_afrag(Arow, 0, hi, aP);
#pragma unroll
    for (int j = 0; j < 4; ++j) load_bfrag(Bcol[j], 0, hi, bP[j]);

    int s = 0;
    while (true) {
        // ---- stage using P buffers; prefetch s+1 into Q ----
        if (s + 2 < S) __builtin_prefetch(Arow + (s + 2) * 32, 0, 0);
        if (s + 1 < S) {
            int kn = (s + 1) * 32;
            load_afrag(Arow, kn, hi, aQ);
#pragma unroll
            for (int j = 0; j < 4; ++j) load_bfrag(Bcol[j], kn, hi, bQ[j]);
        }
#pragma unroll
        for (int j = 0; j < 4; ++j)
            acc[j].v = __builtin_amdgcn_wmma_f32_16x16x32_f16(
                false, aP.v, false, bP[j].v, (short)0, acc[j].v, false, false);
        if (++s >= S) break;

        // ---- stage using Q buffers; prefetch s+1 into P ----
        if (s + 2 < S) __builtin_prefetch(Arow + (s + 2) * 32, 0, 0);
        if (s + 1 < S) {
            int kn = (s + 1) * 32;
            load_afrag(Arow, kn, hi, aP);
#pragma unroll
            for (int j = 0; j < 4; ++j) load_bfrag(Bcol[j], kn, hi, bP[j]);
        }
#pragma unroll
        for (int j = 0; j < 4; ++j)
            acc[j].v = __builtin_amdgcn_wmma_f32_16x16x32_f16(
                false, aQ.v, false, bQ[j].v, (short)0, acc[j].v, false, false);
        if (++s >= S) break;
    }

#pragma unroll
    for (int j = 0; j < 4; ++j) {
        int n  = n_wave + j * 16 + lm;
        int hh = n >> 7, o = n & 127;
        size_t hpb = ((size_t)b * H_ + hh) * N_;
#pragma unroll
        for (int v = 0; v < 8; ++v) {          // C/D: m = hi*8 + v
            int m = m_base + (hi << 3) + v;
            HP[(hpb + m) * O_ + o] = acc[j].f[v];
        }
        size_t tb = (((size_t)b * H_ + hh) * O_ + o) * N_;
#pragma unroll
        for (int v = 0; v < 8; v += 2) {       // packed f16 pair (consecutive m)
            int m = m_base + (hi << 3) + v;
            h2 pp; pp.x = (_Float16)acc[j].f[v]; pp.y = (_Float16)acc[j].f[v + 1];
            *(h2*)(HPT + tb + m) = pp;
        }
    }
}

// ---------------------------------------------------------------------------
// 3. attention logit dots: s/d[b,h,n] = sum_o tanh(hp)*a_{src,dst}[h,o]
// ---------------------------------------------------------------------------
__global__ __launch_bounds__(256) void sd_kernel(
    const float* __restrict__ HP, const float* __restrict__ asrc,
    const float* __restrict__ adst, float* __restrict__ S, float* __restrict__ Dv)
{
    int tid  = threadIdx.x;
    int row  = blockIdx.x * 8 + (tid >> 5);    // (b*H+h)*N + n
    int lane = tid & 31;
    int hh   = (row / N_) % H_;
    const float* base = HP + (size_t)row * O_;
    float s = 0.f, d = 0.f;
#pragma unroll
    for (int t = 0; t < 4; ++t) {
        int o = lane + t * 32;
        float tt = tanhf(base[o]);
        s += tt * asrc[hh * O_ + o];
        d += tt * adst[hh * O_ + o];
    }
    for (int m = 16; m >= 1; m >>= 1) {
        s += __shfl_xor(s, m, 32);
        d += __shfl_xor(d, m, 32);
    }
    if (lane == 0) { S[row] = s; Dv[row] = d; }
}

// ---------------------------------------------------------------------------
// 4. fused attention: leaky-relu + mask + softmax (LDS) + probs x hp (WMMA)
//    global B-operand ping-pong pipelined; A comes from LDS
// ---------------------------------------------------------------------------
__global__ __launch_bounds__(256) void attn_kernel(
    const float* __restrict__ S, const float* __restrict__ Dv,
    const float* __restrict__ adj, const _Float16* __restrict__ HPT,
    _Float16* __restrict__ X1, float* __restrict__ Y, int mode)
{
    __shared__ float    sc[16][N_];       // raw masked logits
    __shared__ _Float16 pr[16][N_];       // unnormalized exp probs (A-operand)
    __shared__ float    rowinv[16];

    int tid = threadIdx.x;
    int b = blockIdx.z, hh = blockIdx.y;
    int bh = b * H_ + hh;
    int r = tid >> 4, c = tid & 15;
    int i = blockIdx.x * 16 + r;

    float si = S[(size_t)bh * N_ + i];
    const float* adjrow = adj + ((size_t)b * N_ + i) * N_;
    const float* Drow   = Dv + (size_t)bh * N_;
    const float NEG_INF = -__builtin_inff();

    float mx = NEG_INF;
    for (int t = 0; t < 32; ++t) {
        int j = c + (t << 4);
        float e = si + Drow[j];
        e = e >= 0.f ? e : 0.2f * e;                  // leaky_relu(0.2)
        e = adjrow[j] > 0.f ? e : NEG_INF;            // adjacency mask
        sc[r][j] = e;
        mx = fmaxf(mx, e);
    }
    for (int m = 8; m >= 1; m >>= 1) mx = fmaxf(mx, __shfl_xor(mx, m, 32));
    float lsum = 0.f;
    for (int t = 0; t < 32; ++t) {
        int j = c + (t << 4);
        float p = __expf(sc[r][j] - mx);
        pr[r][j] = (_Float16)p;
        lsum += p;
    }
    for (int m = 8; m >= 1; m >>= 1) lsum += __shfl_xor(lsum, m, 32);
    if (c == 0) rowinv[r] = 1.0f / lsum;
    __syncthreads();

    // probs[16x512] x hp^T -> out[16x16] per wave, B pipelined
    int wave = tid >> 5, lane = tid & 31, lm = lane & 15, hi = lane >> 4;
    int o = wave * 16 + lm;
    const _Float16* Bcol = HPT + ((size_t)bh * O_ + o) * N_;
    const _Float16* Aprow = &pr[lm][0];

    CF acc;
#pragma unroll
    for (int q = 0; q < 8; ++q) acc.f[q] = 0.f;

    constexpr int S16 = N_ / 32;          // 16 stages
    AF bP, bQ;
    load_bfrag(Bcol, 0, hi, bP);
    int s = 0;
    while (true) {
        if (s + 1 < S16) load_bfrag(Bcol, (s + 1) * 32, hi, bQ);
        { AF a; load_afrag(Aprow, s * 32, hi, a);
          acc.v = __builtin_amdgcn_wmma_f32_16x16x32_f16(
              false, a.v, false, bP.v, (short)0, acc.v, false, false); }
        if (++s >= S16) break;
        if (s + 1 < S16) load_bfrag(Bcol, (s + 1) * 32, hi, bP);
        { AF a; load_afrag(Aprow, s * 32, hi, a);
          acc.v = __builtin_amdgcn_wmma_f32_16x16x32_f16(
              false, a.v, false, bQ.v, (short)0, acc.v, false, false); }
        if (++s >= S16) break;
    }

#pragma unroll
    for (int v = 0; v < 8; ++v) {
        int rr = (hi << 3) + v;
        int node = blockIdx.x * 16 + rr;
        float val = acc.f[v] * rowinv[rr];
        if (mode == 0) {
            float ev = val > 0.f ? val : expm1f(val);   // elu
            X1[((size_t)b * N_ + node) * NOUT_ + hh * O_ + o] = (_Float16)ev;
        } else {
            Y[((size_t)bh * N_ + node) * O_ + o] = val;
        }
    }
}

// ---------------------------------------------------------------------------
// 5. mean over heads + log_softmax over O=128 (one block per (b,n))
// ---------------------------------------------------------------------------
__global__ __launch_bounds__(128) void out_kernel(
    const float* __restrict__ Y, float* __restrict__ out)
{
    int row = blockIdx.x;            // b*N + n
    int b = row >> 9, n = row & 511;
    int o = threadIdx.x;
    float v = 0.f;
#pragma unroll
    for (int hh = 0; hh < H_; ++hh)
        v += Y[(((size_t)b * H_ + hh) * N_ + n) * O_ + o];
    v *= (1.0f / H_);
    __shared__ float buf[128];
    buf[o] = v; __syncthreads();
    for (int s = 64; s > 0; s >>= 1) {
        if (o < s) buf[o] = fmaxf(buf[o], buf[o + s]);
        __syncthreads();
    }
    float mx = buf[0]; __syncthreads();
    float e = __expf(v - mx);
    buf[o] = e; __syncthreads();
    for (int s = 64; s > 0; s >>= 1) {
        if (o < s) buf[o] += buf[o + s];
        __syncthreads();
    }
    float lse = logf(buf[0]);
    out[(size_t)row * O_ + o] = v - mx - lse;
}

// ---------------------------------------------------------------------------
extern "C" void kernel_launch(void* const* d_in, const int* in_sizes, int n_in,
                              void* d_out, int out_size, void* d_ws, size_t ws_size,
                              hipStream_t stream)
{
    (void)in_sizes; (void)n_in; (void)out_size; (void)ws_size;
    const int*   vertices = (const int*)  d_in[0];
    const float* adj      = (const float*)d_in[1];
    const float* hfeat    = (const float*)d_in[2];
    const float* user_emb = (const float*)d_in[3];
    const float* emb_tab  = (const float*)d_in[4];
    const float* n1w = (const float*)d_in[5];
    const float* n1b = (const float*)d_in[6];
    const float* n2w = (const float*)d_in[7];
    const float* n2b = (const float*)d_in[8];
    const float* w0  = (const float*)d_in[9];
    const float* as0 = (const float*)d_in[10];
    const float* ad0 = (const float*)d_in[11];
    const float* w1  = (const float*)d_in[12];
    const float* as1 = (const float*)d_in[13];
    const float* ad1 = (const float*)d_in[14];

    char* ws = (char*)d_ws;
    size_t off = 0;
    auto alloc = [&](size_t bytes) -> char* {
        char* p = ws + off;
        off = (off + bytes + 255) & ~(size_t)255;
        return p;
    };
    _Float16* XA  = (_Float16*)alloc((size_t)B_ * N_ * K0P_ * 2);
    _Float16* W0T = (_Float16*)alloc((size_t)NOUT_ * K0P_ * 2);
    _Float16* W1T = (_Float16*)alloc((size_t)NOUT_ * K1_ * 2);
    float*    HP  = (float*)   alloc((size_t)B_ * H_ * N_ * O_ * 4);   // reused L0/L1
    _Float16* HPT = (_Float16*)alloc((size_t)B_ * H_ * N_ * O_ * 2);   // reused L0/L1
    float*    Sb  = (float*)   alloc((size_t)B_ * H_ * N_ * 4);
    float*    Db  = (float*)   alloc((size_t)B_ * H_ * N_ * 4);
    _Float16* X1  = (_Float16*)alloc((size_t)B_ * N_ * NOUT_ * 2);
    float*    Y   = (float*)   alloc((size_t)B_ * H_ * N_ * O_ * 4);

    // preprocessing / packing
    pack_xh_kernel  <<<(B_ * N_ * K0P_ + 255) / 256, 256, 0, stream>>>(hfeat, XA);
    norm_pack_kernel<<<B_ * (DE_ + UE_),             256, 0, stream>>>(
        vertices, user_emb, emb_tab, n1w, n1b, n2w, n2b, XA);
    pack_w_kernel   <<<(NOUT_ * K0P_ + 255) / 256,   256, 0, stream>>>(w0, W0T, K0P_, K0V_);
    pack_w_kernel   <<<(NOUT_ * K1_  + 255) / 256,   256, 0, stream>>>(w1, W1T, K1_, K1_);

    // layer 0
    gemm_kernel<K0P_><<<dim3(N_ / 64, NOUT_ / 128, B_), 256, 0, stream>>>(XA, W0T, HP, HPT);
    sd_kernel        <<<B_ * H_ * N_ / 8,                256, 0, stream>>>(HP, as0, ad0, Sb, Db);
    attn_kernel      <<<dim3(N_ / 16, H_, B_),           256, 0, stream>>>(Sb, Db, adj, HPT, X1, Y, 0);

    // layer 1 (reuses HP/HPT/Sb/Db)
    gemm_kernel<K1_><<<dim3(N_ / 64, NOUT_ / 128, B_), 256, 0, stream>>>(X1, W1T, HP, HPT);
    sd_kernel       <<<B_ * H_ * N_ / 8,                256, 0, stream>>>(HP, as1, ad1, Sb, Db);
    attn_kernel     <<<dim3(N_ / 16, H_, B_),           256, 0, stream>>>(Sb, Db, adj, HPT, X1, Y, 1);

    // head-mean + log_softmax
    out_kernel<<<B_ * N_, 128, 0, stream>>>(Y, (float*)d_out);
}
